// DSGPM_TP_61280593380081
// MI455X (gfx1250) — compile-verified, matrix operand order checked by reading, not run on hardware
//
#include <hip/hip_runtime.h>
#include <hip/hip_bf16.h>
#include <math.h>

// ---------------------------------------------------------------------------
// Types for CDNA5 wave32 WMMA
// ---------------------------------------------------------------------------
typedef _Float16 h16;
typedef __attribute__((ext_vector_type(8)))  _Float16 v8h;
typedef __attribute__((ext_vector_type(16))) _Float16 v16h;
typedef __attribute__((ext_vector_type(8)))  float    v8f;

// pointer types for the gfx1250 async-to-LDS builtin:
// param0: int4 addrspace(1)* (global source), param1: int4 addrspace(3)* (LDS dest)
typedef int v4i_ __attribute__((vector_size(16)));
typedef __attribute__((address_space(1))) v4i_ g_v4i;
typedef __attribute__((address_space(3))) v4i_ l_v4i;

static __device__ inline v16h cat16(v8h lo, v8h hi) {
    return __builtin_shufflevector(lo, hi, 0,1,2,3,4,5,6,7,8,9,10,11,12,13,14,15);
}

static __device__ inline v8f wmma_f16(v16h a, v16h b, v8f c) {
    // D(f32 16x16) = A(f16 16x32) * B(f16 32x16) + C
    return __builtin_amdgcn_wmma_f32_16x16x32_f16(false, a, false, b, (short)0, c,
                                                  false, false);
}

// K index inside a 16x32 f16 A-fragment for (lane, element i) per ISA layout:
// lanes 0-15: VGPR0-3 -> K=0..7, VGPR4-7 -> K=16..23
// lanes16-31: VGPR0-3 -> K=8..15, VGPR4-7 -> K=24..31
static __device__ __host__ inline int kmap(int lane, int i) {
    return ((i < 8) ? 0 : 16) + ((lane < 16) ? 0 : 8) + (i & 7);
}

// Wait for all outstanding async (LDS-direct) loads issued by this wave.
static __device__ inline void wait_async0() {
#if __has_builtin(__builtin_amdgcn_s_wait_asynccnt)
    __builtin_amdgcn_s_wait_asynccnt(0);
#else
    asm volatile("s_wait_asynccnt 0x0" ::: "memory");
#endif
}

// ---------------------------------------------------------------------------
// Weight repack: W (realK x realN, f32, optionally given transposed as
// realN x realK) -> fragment-ordered f16 pack [kt][nt][lane][i] so each lane
// fetches its 32x16 B-fragment with two contiguous 16B loads.
// ---------------------------------------------------------------------------
__global__ void repack_w(const float* __restrict__ W, h16* __restrict__ out,
                         int realK, int realN, int Kpad, int Npad, int trans) {
    int total = (Kpad >> 5) * (Npad >> 4) * 512;
    int t = blockIdx.x * blockDim.x + threadIdx.x;
    if (t >= total) return;
    int i    = t & 15;
    int lane = (t >> 4) & 31;
    int tile = t >> 9;
    int nNT  = Npad >> 4;
    int nt   = tile % nNT;
    int kt   = tile / nNT;
    int k = kt * 32 + kmap(lane, i);
    int n = nt * 16 + (lane & 15);
    float v = 0.0f;
    if (k < realK && n < realN)
        v = trans ? W[(size_t)n * realK + k] : W[(size_t)k * realN + n];
    out[t] = (h16)v;
}

// ---------------------------------------------------------------------------
// Edge MLP layer 1: A = relu(edge_attr @ We1 + be1)  -> f16  (E x 128)
// ---------------------------------------------------------------------------
__global__ void edge_mlp1(const float* __restrict__ ea, const float* __restrict__ W,
                          const float* __restrict__ b, h16* __restrict__ Af, int E) {
    int t = blockIdx.x * blockDim.x + threadIdx.x;
    if (t >= E * 128) return;
    int e = t >> 7, j = t & 127;
    float s = b[j];
#pragma unroll
    for (int c = 0; c < 4; ++c) s += ea[e * 4 + c] * W[c * 128 + j];
    Af[t] = (h16)fmaxf(s, 0.0f);
}

// ---------------------------------------------------------------------------
// Node init: out0 = concat(relu(embed[idx]), extended_feat) ; h = out0
// ---------------------------------------------------------------------------
__global__ void node_init(const int* __restrict__ x, const float* __restrict__ embed,
                          const float* __restrict__ ext, float* __restrict__ hf,
                          h16* __restrict__ xf, int N) {
    int t = blockIdx.x * blockDim.x + threadIdx.x;
    if (t >= N * 64) return;
    int n = t >> 6, c = t & 63;
    float v;
    if (c < 61) v = fmaxf(embed[x[n] * 61 + c], 0.0f);
    else        v = ext[n * 3 + (c - 61)];
    hf[t] = v;
    xf[t] = (h16)v;
}

__global__ void zero_f32(float* p, long n) {
    long t = (long)blockIdx.x * blockDim.x + threadIdx.x;
    if (t < n) p[t] = 0.0f;
}

// ---------------------------------------------------------------------------
// Generic WMMA GEMM: C = act(A(f16 MxK) @ Bpack(f16 KxNpad) + bias + addend)
// flags: bit0 -> f16 output, bit1 -> relu.
// 128-thread blocks = 4 waves; each wave computes one 16x16 C tile.
// ---------------------------------------------------------------------------
__global__ void gemm_wmma(const h16* __restrict__ A, const h16* __restrict__ Bp,
                          const float* __restrict__ bias, const float* __restrict__ add,
                          void* __restrict__ out, int M, int K, int Npad,
                          int ldC, int storeN, int flags, int totalTiles) {
    int nNT = Npad >> 4;
    int t   = blockIdx.x * 4 + (threadIdx.x >> 5);   // tile id, one per wave
    if (t >= totalTiles) return;                      // wave-uniform exit
    int mt  = t / nNT;
    int nt  = t - mt * nNT;
    int lane = threadIdx.x & 31;
    int lp   = lane & 15;
    int hb   = (lane >> 4) * 8;      // 0 or 8

    int m  = mt * 16 + lp;
    int mA = (m < M) ? m : (M - 1);
    const h16* ar = A + (size_t)mA * K;

    v8f acc = (v8f)0.0f;
    int nKT = K >> 5;
    for (int kt = 0; kt < nKT; ++kt) {
        const h16* ap = ar + kt * 32 + hb;
        v16h af = cat16(*(const v8h*)ap, *(const v8h*)(ap + 16));
        const h16* bp = Bp + (((size_t)kt * nNT + nt) * 32 + lane) * 16;
        v16h bf = cat16(*(const v8h*)bp, *(const v8h*)(bp + 8));
        acc = wmma_f16(af, bf, acc);
    }

    int n = nt * 16 + lp;
    float b = (bias && n < storeN) ? bias[n] : 0.0f;
#pragma unroll
    for (int r = 0; r < 8; ++r) {
        int mr = mt * 16 + r + hb;
        if (mr < M && n < storeN) {
            float v = acc[r] + b;
            if (add) v += add[(size_t)mr * ldC + n];
            if (flags & 2) v = fmaxf(v, 0.0f);
            if (flags & 1) ((h16*)out)[(size_t)mr * ldC + n] = (h16)v;
            else           ((float*)out)[(size_t)mr * ldC + n] = v;
        }
    }
}

// ---------------------------------------------------------------------------
// Edge message kernel (the hot loop).
//   msg_e = (A_e outer o_src) . We2_flat(8192x64) + o_src @ be2_r
//   agg[dst] += msg_e      (f32 atomics)
// 128 threads = 4 waves; each wave owns 16 edges (one M tile).
// A-fragment for fixed j is  A_e[j] * o_src[hchunk*32 + kmap]  -> scalar*vec.
// We2 pack (1 MB) stays L2/WGP$ resident chip-wide.
// Per-block A rows (16 KB) are staged straight into LDS with
// GLOBAL_LOAD_ASYNC_TO_LDS_B128 (ASYNCcnt), drained before the o gathers.
// ---------------------------------------------------------------------------
__global__ void edge_msg_wmma(const h16* __restrict__ Af,   // E x 128 f16
                              const h16* __restrict__ Bp,   // packed 8192x64 f16
                              const h16* __restrict__ xf,   // N x 64 f16 (out)
                              const float* __restrict__ outB, // N x 64 f32: out@be2_r
                              const int* __restrict__ ei,   // edge_index (2,E)
                              float* __restrict__ agg,      // N x 64 f32
                              int E) {
    __shared__ __align__(16) h16 sA[64 * 128];
    __shared__ int sS[64];
    __shared__ int sD[64];

    int tid    = threadIdx.x;
    int wgBase = blockIdx.x * 64;

    if (tid < 64) {
        int e = wgBase + tid; if (e > E - 1) e = E - 1;
        sS[tid] = ei[e];
        sD[tid] = ei[E + e];
    }
    // stage 64 edges' A rows (16 KB) into LDS, coalesced 16B chunks
#if __has_builtin(__builtin_amdgcn_global_load_async_to_lds_b128)
    for (int v = tid; v < 1024; v += 128) {
        int row = v >> 4;
        int e = wgBase + row; if (e > E - 1) e = E - 1;
        void* gp = const_cast<h16*>(Af + (size_t)e * 128 + (size_t)(v & 15) * 8);
        __builtin_amdgcn_global_load_async_to_lds_b128(
            (g_v4i*)gp, (l_v4i*)(sA + (size_t)v * 8), 0, 0);
    }
    wait_async0();
#else
    for (int v = tid; v < 1024; v += 128) {
        int row = v >> 4;
        int e = wgBase + row; if (e > E - 1) e = E - 1;
        *(v8h*)(sA + (size_t)v * 8) =
            *(const v8h*)(Af + (size_t)e * 128 + (size_t)(v & 15) * 8);
    }
#endif
    __syncthreads();

    int wave = tid >> 5;
    int lane = tid & 31;
    int lp   = lane & 15;
    int hb   = (lane >> 4) * 8;
    int eL   = wave * 16 + lp;
    int src  = sS[eL];

    // o-fragments for the two 32-wide h chunks (constant over j)
    const h16* orow = xf + (size_t)src * 64;
    v16h of0 = cat16(*(const v8h*)(orow + hb),      *(const v8h*)(orow + 16 + hb));
    v16h of1 = cat16(*(const v8h*)(orow + 32 + hb), *(const v8h*)(orow + 48 + hb));

    v8f acc[4];
#pragma unroll
    for (int q = 0; q < 4; ++q) acc[q] = (v8f)0.0f;

    for (int j = 0; j < 128; ++j) {
        h16 a = sA[eL * 128 + j];
        v16h af0 = of0 * a;            // k-chunk = j*64 + [0..31]
        v16h af1 = of1 * a;            // k-chunk = j*64 + [32..63]
        size_t kt0 = (size_t)(2 * j) * 4;
#pragma unroll
        for (int nt = 0; nt < 4; ++nt) {
            const h16* bp = Bp + ((kt0 + nt) * 32 + lane) * 16;
            v16h bf = cat16(*(const v8h*)bp, *(const v8h*)(bp + 8));
            acc[nt] = wmma_f16(af0, bf, acc[nt]);
        }
        size_t kt1 = kt0 + 4;
#pragma unroll
        for (int nt = 0; nt < 4; ++nt) {
            const h16* bp = Bp + ((kt1 + nt) * 32 + lane) * 16;
            v16h bf = cat16(*(const v8h*)bp, *(const v8h*)(bp + 8));
            acc[nt] = wmma_f16(af1, bf, acc[nt]);
        }
    }

    // scatter-add: agg[dst] += acc + outB[src]
#pragma unroll
    for (int nt = 0; nt < 4; ++nt) {
        int n = nt * 16 + lp;
#pragma unroll
        for (int r = 0; r < 8; ++r) {
            int m  = r + hb;
            int eg = wgBase + wave * 16 + m;
            if (eg < E) {
                int sl = wave * 16 + m;
                float val = acc[nt][r] + outB[(size_t)sS[sl] * 64 + n];
                unsafeAtomicAdd(agg + (size_t)sD[sl] * 64 + n, val);
            }
        }
    }
}

// ---------------------------------------------------------------------------
// GRU pointwise update
// ---------------------------------------------------------------------------
__global__ void gru_update(const float* __restrict__ gi, const float* __restrict__ gh,
                           float* __restrict__ hf, h16* __restrict__ xf, int N) {
    int t = blockIdx.x * blockDim.x + threadIdx.x;
    if (t >= N * 64) return;
    int n = t >> 6, c = t & 63;
    const float* gin = gi + (size_t)n * 192;
    const float* ghn = gh + (size_t)n * 192;
    float r  = 1.0f / (1.0f + expf(-(gin[c]       + ghn[c])));
    float z  = 1.0f / (1.0f + expf(-(gin[64 + c]  + ghn[64 + c])));
    float nn = tanhf(gin[128 + c] + r * ghn[128 + c]);
    float h  = hf[t];
    h = (1.0f - z) * nn + z * h;
    hf[t] = h;
    xf[t] = (h16)h;
}

// ---------------------------------------------------------------------------
// Build feat = [out2(64) | onehot(24) | ext(3)], L2-normalize.
// Writes fg_embed (N x 91 f32, output 0) and padded f16 copy (N x 96) for the
// classifier GEMM.
// ---------------------------------------------------------------------------
__global__ void feat_norm(const float* __restrict__ out2, const int* __restrict__ x,
                          const float* __restrict__ ext, float* __restrict__ fg,
                          h16* __restrict__ featp, int N) {
    int n = blockIdx.x * blockDim.x + threadIdx.x;
    if (n >= N) return;
    const float* o = out2 + (size_t)n * 64;
    int id = x[n];
    float e0 = ext[n * 3], e1 = ext[n * 3 + 1], e2 = ext[n * 3 + 2];
    float ss = 1.0f + e0 * e0 + e1 * e1 + e2 * e2;   // onehot contributes 1
    for (int c = 0; c < 64; ++c) ss += o[c] * o[c];
    float inv = 1.0f / fmaxf(sqrtf(ss), 1e-12f);
    float* fr = fg + (size_t)n * 91;
    h16*   fp = featp + (size_t)n * 96;
    for (int c = 0; c < 64; ++c) { float v = o[c] * inv; fr[c] = v; fp[c] = (h16)v; }
    for (int a = 0; a < 24; ++a) {
        float v = (a == id) ? inv : 0.0f;
        fr[64 + a] = v; fp[64 + a] = (h16)v;
    }
    fr[88] = e0 * inv; fp[88] = (h16)(e0 * inv);
    fr[89] = e1 * inv; fp[89] = (h16)(e1 * inv);
    fr[90] = e2 * inv; fp[90] = (h16)(e2 * inv);
    for (int c = 91; c < 96; ++c) fp[c] = (h16)0.0f;
}

// ---------------------------------------------------------------------------
// Host orchestration
// ---------------------------------------------------------------------------
static inline int gemm_tiles(int M, int Npad) { return ((M + 15) / 16) * (Npad / 16); }

static inline void launch_gemm(hipStream_t stream, const h16* A, const h16* Bp,
                               const float* bias, const float* add, void* out,
                               int M, int K, int Npad, int ldC, int storeN, int flags) {
    int tiles = gemm_tiles(M, Npad);
    gemm_wmma<<<(tiles + 3) / 4, 128, 0, stream>>>(A, Bp, bias, add, out, M, K, Npad,
                                                   ldC, storeN, flags, tiles);
}

extern "C" void kernel_launch(void* const* d_in, const int* in_sizes, int n_in,
                              void* d_out, int out_size, void* d_ws, size_t ws_size,
                              hipStream_t stream) {
    const int*   x     = (const int*)  d_in[0];
    const int*   ei    = (const int*)  d_in[1];
    const float* ea    = (const float*)d_in[2];
    const float* ext   = (const float*)d_in[3];
    const float* embed = (const float*)d_in[4];
    const float* We1   = (const float*)d_in[5];
    const float* be1   = (const float*)d_in[6];
    const float* We2   = (const float*)d_in[7];
    const float* be2   = (const float*)d_in[8];
    const float* root  = (const float*)d_in[9];
    const float* convb = (const float*)d_in[10];
    const float* Wih   = (const float*)d_in[11];
    const float* Whh   = (const float*)d_in[12];
    const float* bih   = (const float*)d_in[13];
    const float* bhh   = (const float*)d_in[14];
    const float* Wo1   = (const float*)d_in[15];
    const float* bo1   = (const float*)d_in[16];
    const float* Wo2   = (const float*)d_in[17];
    const float* bo2   = (const float*)d_in[18];
    const float* Wc1   = (const float*)d_in[19];
    const float* bc1   = (const float*)d_in[20];
    const float* Wc2   = (const float*)d_in[21];
    const float* bc2   = (const float*)d_in[22];

    const int N = in_sizes[0];
    const int E = in_sizes[1] / 2;

    // bump allocator over workspace
    size_t off = 0;
    auto alloc = [&](size_t bytes) -> void* {
        void* p = (char*)d_ws + off;
        off = (off + bytes + 255) & ~(size_t)255;
        return p;
    };
    h16*   Af16   = (h16*)  alloc((size_t)E * 128 * 2);
    h16*   bpWe2  = (h16*)  alloc((size_t)(8192 / 32) * (64 / 16) * 512 * 2);
    h16*   bpBe2  = (h16*)  alloc((size_t)2 * 4 * 512 * 2);
    h16*   bpRoot = (h16*)  alloc((size_t)2 * 4 * 512 * 2);
    h16*   bpWih  = (h16*)  alloc((size_t)2 * 12 * 512 * 2);
    h16*   bpWhh  = (h16*)  alloc((size_t)2 * 12 * 512 * 2);
    h16*   bpWo1  = (h16*)  alloc((size_t)2 * 4 * 512 * 2);
    h16*   bpWo2  = (h16*)  alloc((size_t)2 * 4 * 512 * 2);
    h16*   bpWc1  = (h16*)  alloc((size_t)3 * 16 * 512 * 2);
    h16*   bpWc2  = (h16*)  alloc((size_t)8 * 2 * 512 * 2);
    h16*   xf16   = (h16*)  alloc((size_t)N * 64 * 2);
    float* hf32   = (float*)alloc((size_t)N * 64 * 4);
    float* aggf   = (float*)alloc((size_t)N * 64 * 4);
    float* outBf  = (float*)alloc((size_t)N * 64 * 4);   // reused as out2 at the end
    h16*   mf16   = (h16*)  alloc((size_t)N * 64 * 2);   // reused as t1 at the end
    float* gif    = (float*)alloc((size_t)N * 192 * 4);
    float* ghf    = (float*)alloc((size_t)N * 192 * 4);
    h16*   featp  = (h16*)gif;                            // N x 96 f16, reuse
    h16*   c1f16  = (h16*)ghf;                            // N x 256 f16, reuse
    (void)ws_size; (void)n_in; (void)out_size;

    float* fg   = (float*)d_out;                  // N x 91
    float* pred = (float*)d_out + (size_t)N * 91; // N x 18

    const int TPB = 256;

    // ---- one-time setup ----
    repack_w<<<(524288 + TPB - 1) / TPB, TPB, 0, stream>>>(We2, bpWe2, 8192, 64, 8192, 64, 0);
    repack_w<<<(2 * 4 * 512 + TPB - 1) / TPB, TPB, 0, stream>>>(be2,  bpBe2, 64, 64, 64, 64, 0);
    repack_w<<<(2 * 4 * 512 + TPB - 1) / TPB, TPB, 0, stream>>>(root, bpRoot, 64, 64, 64, 64, 0);
    repack_w<<<(2 * 12 * 512 + TPB - 1) / TPB, TPB, 0, stream>>>(Wih, bpWih, 64, 192, 64, 192, 1);
    repack_w<<<(2 * 12 * 512 + TPB - 1) / TPB, TPB, 0, stream>>>(Whh, bpWhh, 64, 192, 64, 192, 1);
    repack_w<<<(2 * 4 * 512 + TPB - 1) / TPB, TPB, 0, stream>>>(Wo1, bpWo1, 64, 64, 64, 64, 0);
    repack_w<<<(2 * 4 * 512 + TPB - 1) / TPB, TPB, 0, stream>>>(Wo2, bpWo2, 64, 64, 64, 64, 0);
    repack_w<<<(3 * 16 * 512 + TPB - 1) / TPB, TPB, 0, stream>>>(Wc1, bpWc1, 91, 256, 96, 256, 0);
    repack_w<<<(8 * 2 * 512 + TPB - 1) / TPB, TPB, 0, stream>>>(Wc2, bpWc2, 256, 18, 256, 32, 0);

    edge_mlp1<<<((size_t)E * 128 + TPB - 1) / TPB, TPB, 0, stream>>>(ea, We1, be1, Af16, E);
    node_init<<<((size_t)N * 64 + TPB - 1) / TPB, TPB, 0, stream>>>(x, embed, ext, hf32, xf16, N);

    // ---- message-passing iterations ----
    const int edgeBlocks = (E + 63) / 64;
    for (int it = 0; it < 6; ++it) {
        zero_f32<<<((size_t)N * 64 + TPB - 1) / TPB, TPB, 0, stream>>>(aggf, (long)N * 64);
        // outB = out @ be2_r
        launch_gemm(stream, xf16, bpBe2, nullptr, nullptr, outBf, N, 64, 64, 64, 64, 0);
        // agg[dst] += (A_e outer o_src) . We2 + outB[src]
        edge_msg_wmma<<<edgeBlocks, 128, 0, stream>>>(Af16, bpWe2, xf16, outBf, ei, aggf, E);
        // m = relu(out @ root + agg + conv_b)
        launch_gemm(stream, xf16, bpRoot, convb, aggf, mf16, N, 64, 64, 64, 64, 1 | 2);
        // gi = m @ Wih^T + bih ; gh = h @ Whh^T + bhh
        launch_gemm(stream, mf16, bpWih, bih, nullptr, gif, N, 64, 192, 192, 192, 0);
        launch_gemm(stream, xf16, bpWhh, bhh, nullptr, ghf, N, 64, 192, 192, 192, 0);
        gru_update<<<((size_t)N * 64 + TPB - 1) / TPB, TPB, 0, stream>>>(gif, ghf, hf32, xf16, N);
    }

    // ---- output head ----
    // t1 = relu(h @ Wo1 + bo1)   (reuse mf16)
    launch_gemm(stream, xf16, bpWo1, bo1, nullptr, mf16, N, 64, 64, 64, 64, 1 | 2);
    // out2 = t1 @ Wo2 + bo2      (reuse outBf)
    launch_gemm(stream, mf16, bpWo2, bo2, nullptr, outBf, N, 64, 64, 64, 64, 0);
    // fg_embed + padded f16 feat
    feat_norm<<<(N + TPB - 1) / TPB, TPB, 0, stream>>>(outBf, x, ext, fg, featp, N);
    // c1 = relu(feat @ Wc1 + bc1)
    launch_gemm(stream, featp, bpWc1, bc1, nullptr, c1f16, N, 96, 256, 256, 256, 1 | 2);
    // pred = c1 @ Wc2 + bc2   (Npad=32, store 18 cols)
    launch_gemm(stream, c1f16, bpWc2, bc2, nullptr, pred, N, 256, 32, 18, 18, 0);
}
